// SAGNetworkHierarchical_22874995818684
// MI455X (gfx1250) — matrix operand order repaired and checked by reading, not compiled, not run on hardware
//
#include <hip/hip_runtime.h>

#define NNODES 50000
#define NEDGES 800000
#define HD 128

typedef __attribute__((ext_vector_type(16))) _Float16 v16h;
typedef __attribute__((ext_vector_type(8)))  _Float16 v8h;
typedef __attribute__((ext_vector_type(8)))  float    v8f;

struct Ctrl {
  unsigned kRemain;
  unsigned prefix;
  unsigned selCounter;
  unsigned hist[256];
};

__device__ __forceinline__ unsigned f2key(float f) {
  unsigned u = __float_as_uint(f);
  return (u & 0x80000000u) ? ~u : (u | 0x80000000u);
}
__device__ __forceinline__ float key2f(unsigned k) {
  unsigned u = (k & 0x80000000u) ? (k & 0x7FFFFFFFu) : ~k;
  return __uint_as_float(u);
}

// ---------------- generic fills ----------------
__global__ void fill_f32(float* p, float v, int n) {
  int i = blockIdx.x * blockDim.x + threadIdx.x;
  if (i < n) p[i] = v;
}
__global__ void fill_u32(unsigned* p, unsigned v, int n) {
  int i = blockIdx.x * blockDim.x + threadIdx.x;
  if (i < n) p[i] = v;
}

__global__ void init_edges_k(const int* __restrict__ s, const int* __restrict__ d,
                             int* sc, int* dc, float* em, int E) {
  int i = blockIdx.x * blockDim.x + threadIdx.x;
  if (i < E) { sc[i] = s[i]; dc[i] = d[i]; em[i] = 1.0f; }
}

// ---------------- WMMA GEMM: Y[n,128] = X[n,128] @ W[128,128] ----------------
// One workgroup (8 waves) per 16-row strip; wave w owns 16x16 tile at col 16*w.
// A staged row-major, B staged TRANSPOSED (col-major) so every lane fragment is
// two contiguous 16B chunks -> ds_load_b128 pairs, no half-register repacking.
// Full-tile stores take a uniform unguarded fast path (only the final ragged
// workgroup per launch runs the per-row guarded path).
__global__ __launch_bounds__(256) void gemm_wmma_k(const float* __restrict__ X,
                                                   const float* __restrict__ W,
                                                   float* __restrict__ Y, int n) {
  __shared__ _Float16 As[16 * 136];    // [row][k], stride 136 (272B = 17*16, aligned)
  __shared__ _Float16 Bt[128 * 136];   // [col][k], transposed W
  const int tid = threadIdx.x;
  const int m0  = blockIdx.x * 16;

  for (int idx = tid; idx < 16 * 128; idx += 256) {
    int r = idx >> 7, cc = idx & 127;
    int m = m0 + r;
    As[r * 136 + cc] = (m < n) ? (_Float16)X[(size_t)m * HD + cc] : (_Float16)0.0f;
  }
  for (int idx = tid; idx < 128 * 128; idx += 256) {
    int r = idx >> 7, cc = idx & 127;           // W[r][cc], coalesced global read
    Bt[cc * 136 + r] = (_Float16)W[idx];        // transposed LDS write
  }
  __syncthreads();

  const int wave  = tid >> 5;
  const int lane  = tid & 31;
  const int ncol0 = wave * 16;
  const int hi    = lane >> 4;          // lane half selects K sub-ranges
  const int rsel  = lane & 15;
  const _Float16* arow = &As[rsel * 136];
  const _Float16* brow = &Bt[(ncol0 + rsel) * 136];

  v8f c = {};
  for (int k0 = 0; k0 < 128; k0 += 32) {
    // A (16-bit 16x32 layout): lane half 0 -> K+0..7 & K+16..23, half 1 -> +8
    const int ka = k0 + hi * 8;
    v8h a0 = *(const v8h*)(arow + ka);
    v8h a1 = *(const v8h*)(arow + ka + 16);
    v16h a = __builtin_shufflevector(a0, a1, 0, 1, 2, 3, 4, 5, 6, 7,
                                     8, 9, 10, 11, 12, 13, 14, 15);
    // B (32x16): lane half 0 -> K+0..7 & K+16..23, half 1 -> +8 (matches A)
    const int kb = k0 + hi * 8;
    v8h b0 = *(const v8h*)(brow + kb);
    v8h b1 = *(const v8h*)(brow + kb + 16);
    v16h b = __builtin_shufflevector(b0, b1, 0, 1, 2, 3, 4, 5, 6, 7,
                                     8, 9, 10, 11, 12, 13, 14, 15);
    c = __builtin_amdgcn_wmma_f32_16x16x32_f16(false, a, false, b, (short)0, c,
                                               false, false);
  }

  // D: VGPR r -> rows m0 + r + hi*8, col ncol0 + rsel
  float* ybase = Y + (size_t)(m0 + hi * 8) * HD + ncol0 + rsel;
  if (m0 + 16 <= n) {                   // uniform: full tile, straight-line stores
#pragma unroll
    for (int r = 0; r < 8; ++r) ybase[(size_t)r * HD] = c[r];
  } else {                              // ragged tail block only
#pragma unroll
    for (int r = 0; r < 8; ++r) {
      int m = m0 + r + hi * 8;
      if (m < n) ybase[(size_t)r * HD] = c[r];
    }
  }
}

// ---------------- graph message passing ----------------
__global__ void degree_k(const int* __restrict__ src, const int* __restrict__ dst,
                         const float* __restrict__ em, float* degO, float* degI, int E) {
  int e = blockIdx.x * blockDim.x + threadIdx.x;
  if (e >= E) return;
  float m = em[e];
  if (m == 0.0f) return;
  unsafeAtomicAdd(&degO[src[e]], m);
  unsafeAtomicAdd(&degI[dst[e]], m);
}

__global__ void invsqrt_k(float* d, int n) {
  int i = blockIdx.x * blockDim.x + threadIdx.x;
  if (i < n) d[i] = rsqrtf(fmaxf(d[i], 1.0f));
}

// 32 threads per edge, 4 floats per thread: coalesced 512B gather, L2 fp atomics scatter
__global__ void msg_feat_k(const float* __restrict__ hlin, const int* __restrict__ src,
                           const int* __restrict__ dst, const float* __restrict__ em,
                           const float* __restrict__ ns, float* agg, int E) {
  int t = blockIdx.x * blockDim.x + threadIdx.x;
  int e = t >> 5;
  if (e >= E) return;
  float m = em[e];
  if (m == 0.0f) return;
  int s = src[e], d = dst[e];
  float w = m * ns[s];
  int f = (t & 31) << 2;
  const float4 v = *(const float4*)(hlin + (size_t)s * HD + f);
  float* ap = agg + (size_t)d * HD + f;
  unsafeAtomicAdd(ap + 0, v.x * w);
  unsafeAtomicAdd(ap + 1, v.y * w);
  unsafeAtomicAdd(ap + 2, v.z * w);
  unsafeAtomicAdd(ap + 3, v.w * w);
}

__global__ void finish_conv_k(float* h, const float* __restrict__ nd,
                              const float* __restrict__ bias, int n) {
  int i = blockIdx.x * blockDim.x + threadIdx.x;
  if (i >= n * HD) return;
  int node = i >> 7, f = i & 127;
  float v = h[i] * nd[node] + bias[f];
  h[i] = fmaxf(v, 0.0f);
}

// per-node dot h[i,:] . sW  (one wave per node)
__global__ void score_dot_k(const float* __restrict__ h, const float* __restrict__ sW,
                            float* hs, int n) {
  int g = blockIdx.x * blockDim.x + threadIdx.x;
  int node = g >> 5;
  int lane = threadIdx.x & 31;
  if (node >= n) return;
  const float* row = h + (size_t)node * HD;
  float s = 0.0f;
  for (int j = lane; j < HD; j += 32) s += row[j] * sW[j];
  for (int off = 16; off; off >>= 1) s += __shfl_xor(s, off, 32);
  if (lane == 0) hs[node] = s;
}

__global__ void msg_scalar_k(const float* __restrict__ hs, const int* __restrict__ src,
                             const int* __restrict__ dst, const float* __restrict__ em,
                             const float* __restrict__ ns, float* aggs, int E) {
  int e = blockIdx.x * blockDim.x + threadIdx.x;
  if (e >= E) return;
  float m = em[e];
  if (m == 0.0f) return;
  int s = src[e];
  unsafeAtomicAdd(&aggs[dst[e]], hs[s] * ns[s] * m);
}

__global__ void finish_score_k(const float* __restrict__ aggs, const float* __restrict__ nd,
                               const float* __restrict__ sb, float* score, unsigned* keys,
                               int n) {
  int i = blockIdx.x * blockDim.x + threadIdx.x;
  if (i >= n) return;
  float s = aggs[i] * nd[i] + sb[0];
  score[i] = s;
  keys[i] = f2key(s);
}

// ---------------- device-side exact radix select (k-th largest key) ----------------
__global__ void init_select_k(Ctrl* c, int k) {
  if (threadIdx.x == 0) {
    c->kRemain = (unsigned)k;
    c->prefix = 0u;
    c->selCounter = 0u;
    for (int j = 0; j < 256; ++j) c->hist[j] = 0u;
  }
}

__global__ void hist_k(const unsigned* __restrict__ keys, int n, int byt, Ctrl* c) {
  int i = blockIdx.x * blockDim.x + threadIdx.x;
  if (i >= n) return;
  unsigned key = keys[i];
  unsigned maskAbove = (byt == 3) ? 0u : (0xFFFFFFFFu << ((byt + 1) * 8));
  if ((key & maskAbove) == (c->prefix & maskAbove))
    atomicAdd(&c->hist[(key >> (byt * 8)) & 0xFFu], 1u);
}

__global__ void scan_k(Ctrl* c, int byt) {
  if (threadIdx.x != 0) return;
  unsigned kr = c->kRemain;
  unsigned cum = 0;
  for (int bin = 255; bin >= 0; --bin) {
    unsigned cnt = c->hist[bin];
    if (cum + cnt >= kr) {
      c->prefix |= ((unsigned)bin) << (byt * 8);
      c->kRemain = kr - cum;
      break;
    }
    cum += cnt;
  }
  for (int j = 0; j < 256; ++j) c->hist[j] = 0u;
}

__global__ void mark_k(const unsigned* __restrict__ keys, int n, Ctrl* c, int* newid) {
  int i = blockIdx.x * blockDim.x + threadIdx.x;
  if (i >= n) return;
  if (keys[i] > c->prefix) newid[i] = (int)atomicAdd(&c->selCounter, 1u);
}

// deterministic lowest-index tie inclusion, single wave sweep with ballot
__global__ void tie_k(const unsigned* __restrict__ keys, int n, Ctrl* c, int* newid) {
  const unsigned T = c->prefix;
  const int need = (int)c->kRemain;
  const int lane = threadIdx.x;
  int taken = 0;
  for (int base = 0; base < n && taken < need; base += 32) {
    int i = base + lane;
    bool tie = (i < n) && (keys[i] == T);
    unsigned m32 = (unsigned)__ballot(tie);
    int before = __popc(m32 & ((1u << lane) - 1u));
    if (tie && (taken + before) < need)
      newid[i] = (int)atomicAdd(&c->selCounter, 1u);
    taken += __popc(m32);
  }
}

// ---------------- pool + readout + relabel ----------------
__global__ void pool_k(const float* __restrict__ h, const float* __restrict__ score,
                       const int* __restrict__ newid, float* xnext, float* rosum,
                       unsigned* romax, int n) {
  int t = blockIdx.x * blockDim.x + threadIdx.x;
  int node = t >> 5;
  if (node >= n) return;
  int nid = newid[node];
  if (nid < 0) return;
  int f = (t & 31) << 2;
  float g = tanhf(score[node]);
  float4 v = *(const float4*)(h + (size_t)node * HD + f);
  v.x *= g; v.y *= g; v.z *= g; v.w *= g;
  *(float4*)(xnext + (size_t)nid * HD + f) = v;
  unsafeAtomicAdd(rosum + f + 0, v.x);
  unsafeAtomicAdd(rosum + f + 1, v.y);
  unsafeAtomicAdd(rosum + f + 2, v.z);
  unsafeAtomicAdd(rosum + f + 3, v.w);
  atomicMax(romax + f + 0, f2key(v.x));
  atomicMax(romax + f + 1, f2key(v.y));
  atomicMax(romax + f + 2, f2key(v.z));
  atomicMax(romax + f + 3, f2key(v.w));
}

__global__ void combine_k(const float* __restrict__ rosum, const unsigned* __restrict__ romax,
                          float* rofinal, float invk) {
  int f = threadIdx.x;  // 128 threads
  rofinal[f]       += rosum[f] * invk;
  rofinal[128 + f] += key2f(romax[f]);
}

__global__ void remap_k(const int* __restrict__ newid, int* sc, int* dc, float* em, int E) {
  int e = blockIdx.x * blockDim.x + threadIdx.x;
  if (e >= E) return;
  int nsv = newid[sc[e]];
  int ndv = newid[dc[e]];
  float m = em[e];
  if (nsv < 0 || ndv < 0) m = 0.0f;
  sc[e] = nsv < 0 ? 0 : nsv;
  dc[e] = ndv < 0 ? 0 : ndv;
  em[e] = m;
}

// ---------------- MLP head ----------------
__global__ void build_final_k(const float* __restrict__ rofinal,
                              const float* __restrict__ seq, float* fin) {
  int i = blockIdx.x * blockDim.x + threadIdx.x;
  if (i >= 1280) return;
  fin[i] = (i < 256) ? rofinal[i] : seq[i - 256];
}

__global__ void linear_k(const float* __restrict__ v, const float* __restrict__ W,
                         const float* __restrict__ b, float* out, int indim, int outdim,
                         int dorelu) {
  int j = blockIdx.x * blockDim.x + threadIdx.x;
  if (j >= outdim) return;
  float acc = b[j];
  for (int i = 0; i < indim; ++i) acc += v[i] * W[(size_t)i * outdim + j];
  if (dorelu) acc = fmaxf(acc, 0.0f);
  out[j] = acc;
}

// ---------------- host side ----------------
extern "C" void kernel_launch(void* const* d_in, const int* in_sizes, int n_in,
                              void* d_out, int out_size, void* d_ws, size_t ws_size,
                              hipStream_t stream) {
  (void)in_sizes; (void)n_in; (void)out_size; (void)ws_size;
  const float* feat = (const float*)d_in[0];
  const float* seq  = (const float*)d_in[1];
  const int*   srcI = (const int*)d_in[2];
  const int*   dstI = (const int*)d_in[3];
  // d_in[4], d_in[5] (label edges) and d_in[24..26] (GAT params): dead code in reference
  const float* cW[3] = {(const float*)d_in[6], (const float*)d_in[10], (const float*)d_in[14]};
  const float* cb[3] = {(const float*)d_in[7], (const float*)d_in[11], (const float*)d_in[15]};
  const float* sW[3] = {(const float*)d_in[8], (const float*)d_in[12], (const float*)d_in[16]};
  const float* sb[3] = {(const float*)d_in[9], (const float*)d_in[13], (const float*)d_in[17]};
  const float* l1W = (const float*)d_in[18];
  const float* l1b = (const float*)d_in[19];
  const float* l2W = (const float*)d_in[20];
  const float* l2b = (const float*)d_in[21];
  const float* l3W = (const float*)d_in[22];
  const float* l3b = (const float*)d_in[23];
  float* out = (float*)d_out;

  char* ws = (char*)d_ws;
  size_t off = 0;
  auto alloc = [&](size_t bytes) -> void* {
    void* p = ws + off;
    off = (off + bytes + 255) & ~(size_t)255;
    return p;
  };
  float*    xcur  = (float*)alloc((size_t)NNODES * HD * 4);
  float*    hlin  = (float*)alloc((size_t)NNODES * HD * 4);
  float*    hbuf  = (float*)alloc((size_t)NNODES * HD * 4);  // agg then h (in place)
  float*    degO  = (float*)alloc((size_t)NNODES * 4);
  float*    degI  = (float*)alloc((size_t)NNODES * 4);
  float*    hs    = (float*)alloc((size_t)NNODES * 4);
  float*    aggs  = (float*)alloc((size_t)NNODES * 4);
  float*    score = (float*)alloc((size_t)NNODES * 4);
  unsigned* keys  = (unsigned*)alloc((size_t)NNODES * 4);
  int*      newid = (int*)alloc((size_t)NNODES * 4);
  int*      srcC  = (int*)alloc((size_t)NEDGES * 4);
  int*      dstC  = (int*)alloc((size_t)NEDGES * 4);
  float*    em    = (float*)alloc((size_t)NEDGES * 4);
  float*    rosum = (float*)alloc(128 * 4);
  unsigned* romax = (unsigned*)alloc(128 * 4);
  float*    rofin = (float*)alloc(256 * 4);
  float*    finv  = (float*)alloc(1280 * 4);
  float*    f1    = (float*)alloc(256 * 4);
  float*    f2    = (float*)alloc(128 * 4);
  Ctrl*     ctrl  = (Ctrl*)alloc(sizeof(Ctrl));

  auto cdiv = [](int a, int b) { return (a + b - 1) / b; };
  const unsigned KEY_NEG_INF = 0x007FFFFFu;  // f2key(-inf)

  fill_f32<<<1, 256, 0, stream>>>(rofin, 0.0f, 256);
  init_edges_k<<<cdiv(NEDGES, 256), 256, 0, stream>>>(srcI, dstI, srcC, dstC, em, NEDGES);

  int n = NNODES;
  const float* X = feat;
  for (int b = 0; b < 3; ++b) {
    int k = (n + 1) / 2;  // ceil(0.5*n): 25000, 12500, 6250 (static)
    fill_f32<<<cdiv(n * HD, 256), 256, 0, stream>>>(hbuf, 0.0f, n * HD);
    fill_f32<<<cdiv(n, 256), 256, 0, stream>>>(degO, 0.0f, n);
    fill_f32<<<cdiv(n, 256), 256, 0, stream>>>(degI, 0.0f, n);
    fill_f32<<<cdiv(n, 256), 256, 0, stream>>>(aggs, 0.0f, n);
    fill_u32<<<cdiv(n, 256), 256, 0, stream>>>((unsigned*)newid, 0xFFFFFFFFu, n);
    fill_f32<<<1, 128, 0, stream>>>(rosum, 0.0f, 128);
    fill_u32<<<1, 128, 0, stream>>>(romax, KEY_NEG_INF, 128);

    // GraphConv
    gemm_wmma_k<<<cdiv(n, 16), 256, 0, stream>>>(X, cW[b], hlin, n);
    degree_k<<<cdiv(NEDGES, 256), 256, 0, stream>>>(srcC, dstC, em, degO, degI, NEDGES);
    invsqrt_k<<<cdiv(n, 256), 256, 0, stream>>>(degO, n);
    invsqrt_k<<<cdiv(n, 256), 256, 0, stream>>>(degI, n);
    msg_feat_k<<<cdiv(NEDGES * 32, 256), 256, 0, stream>>>(hlin, srcC, dstC, em, degO, hbuf, NEDGES);
    finish_conv_k<<<cdiv(n * HD, 256), 256, 0, stream>>>(hbuf, degI, cb[b], n);

    // Score GraphConv
    score_dot_k<<<cdiv(n * 32, 256), 256, 0, stream>>>(hbuf, sW[b], hs, n);
    msg_scalar_k<<<cdiv(NEDGES, 256), 256, 0, stream>>>(hs, srcC, dstC, em, degO, aggs, NEDGES);
    finish_score_k<<<cdiv(n, 256), 256, 0, stream>>>(aggs, degI, sb[b], score, keys, n);

    // Exact top-k via 4-pass radix select
    init_select_k<<<1, 1, 0, stream>>>(ctrl, k);
    for (int byt = 3; byt >= 0; --byt) {
      hist_k<<<cdiv(n, 256), 256, 0, stream>>>(keys, n, byt, ctrl);
      scan_k<<<1, 1, 0, stream>>>(ctrl, byt);
    }
    mark_k<<<cdiv(n, 256), 256, 0, stream>>>(keys, n, ctrl, newid);
    tie_k<<<1, 32, 0, stream>>>(keys, n, ctrl, newid);

    // Gate + compact + mean/max readout + edge relabel
    pool_k<<<cdiv(n * 32, 256), 256, 0, stream>>>(hbuf, score, newid, xcur, rosum, romax, n);
    combine_k<<<1, 128, 0, stream>>>(rosum, romax, rofin, 1.0f / (float)k);
    remap_k<<<cdiv(NEDGES, 256), 256, 0, stream>>>(newid, srcC, dstC, em, NEDGES);

    X = xcur;
    n = k;
  }

  build_final_k<<<cdiv(1280, 256), 256, 0, stream>>>(rofin, seq, finv);
  linear_k<<<1, 256, 0, stream>>>(finv, l1W, l1b, f1, 1280, 256, 1);
  linear_k<<<1, 128, 0, stream>>>(f1, l2W, l2b, f2, 256, 128, 1);
  linear_k<<<cdiv(2048, 256), 256, 0, stream>>>(f2, l3W, l3b, out, 128, 2048, 0);
}
